// MoE_87660282511383
// MI455X (gfx1250) — compile-verified, matrix operand order checked by reading, not run on hardware
//
#include <hip/hip_runtime.h>
#include <hip/hip_bf16.h>

// ---- problem constants (from reference) ----
#define N_TOK 4096   // B*T = 2*2048
#define DIM   1024   // D
#define HID   4096   // H
#define NEXP  8      // E

#define PADE  40     // LDS row stride in bf16 elements (80B: banks spread, 16B-aligned chunks)
#define ROWB  (PADE * 2)  // LDS row stride in bytes

typedef __attribute__((ext_vector_type(16))) __bf16 v16bf;
typedef __attribute__((ext_vector_type(8)))  __bf16 v8bf;
typedef __attribute__((ext_vector_type(8)))  float  v8f;

// ---------- helpers ----------
__device__ __forceinline__ unsigned short f32_to_bf16_rne(float f) {
    union { float f; unsigned u; } v; v.f = f;
    unsigned u = v.u;
    unsigned r = u + 0x7FFFu + ((u >> 16) & 1u);
    return (unsigned short)(r >> 16);
}

// one 16-byte async global->LDS copy (CDNA5 async path, tracked by ASYNCcnt)
__device__ __forceinline__ void async_copy16(unsigned lds_addr, const unsigned short* gptr) {
    asm volatile("global_load_async_to_lds_b128 %0, %1, off"
                 :: "v"(lds_addr), "v"(gptr)
                 : "memory");
}
__device__ __forceinline__ void wait_async_le4() {
    asm volatile("s_wait_asynccnt 0x4" ::: "memory");
}
__device__ __forceinline__ void wait_async_0() {
    asm volatile("s_wait_asynccnt 0x0" ::: "memory");
}

__device__ __forceinline__ v16bf frag_from_lds(const unsigned short* p /*row base + hi*8*/) {
    const v8bf* pv = (const v8bf*)p;
    v8bf lo = pv[0];
    v8bf hs = pv[2];   // +16 elements (= +32B) : K+16..K+23 per ISA layout
    return __builtin_shufflevector(lo, hs, 0,1,2,3,4,5,6,7,8,9,10,11,12,13,14,15);
}

// ---------- init ----------
__global__ void moe_zero_kernel(float* __restrict__ out, float* __restrict__ psum,
                                float* __restrict__ cnt, int nOut) {
    int i = blockIdx.x * blockDim.x + threadIdx.x;
    if (i < nOut) out[i] = 0.0f;
    if (i < NEXP) { psum[i] = 0.0f; cnt[i] = 0.0f; }
}

// ---------- fp32 -> bf16 (4-wide) ----------
__global__ void moe_cvt_bf16_kernel(const float* __restrict__ src,
                                    unsigned short* __restrict__ dst, int n4) {
    int i = blockIdx.x * blockDim.x + threadIdx.x;
    if (i >= n4) return;
    const float4 f = ((const float4*)src)[i];
    ushort4 o;
    o.x = f32_to_bf16_rne(f.x); o.y = f32_to_bf16_rne(f.y);
    o.z = f32_to_bf16_rne(f.z); o.w = f32_to_bf16_rne(f.w);
    ((ushort4*)dst)[i] = o;
}

// ---------- gating: logits[n,e] = dot(x[n,:], W_gate[e,:]) in fp32 ----------
__global__ void moe_gate_logits_kernel(const float* __restrict__ x,
                                       const float* __restrict__ Wg,
                                       float* __restrict__ logits) {
    int idx = blockIdx.x * blockDim.x + threadIdx.x;
    if (idx >= N_TOK * NEXP) return;
    int n = idx >> 3, e = idx & 7;
    const float* xr = x  + (size_t)n * DIM;
    const float* wr = Wg + (size_t)e * DIM;
    float acc = 0.0f;
    #pragma unroll 8
    for (int d = 0; d < DIM; ++d) acc = fmaf(xr[d], wr[d], acc);
    logits[idx] = acc;
}

// ---------- gating: softmax, top-2, renormalized routing weights ----------
__global__ void moe_gate_finalize_kernel(const float* __restrict__ logits,
                                         float* __restrict__ we,
                                         float* __restrict__ psum,
                                         float* __restrict__ cnt) {
    int n = blockIdx.x * blockDim.x + threadIdx.x;
    if (n >= N_TOK) return;
    float l[NEXP], p[NEXP];
    float mx = -3.0e38f;
    #pragma unroll
    for (int e = 0; e < NEXP; ++e) { l[e] = logits[n * NEXP + e]; mx = fmaxf(mx, l[e]); }
    float s = 0.0f;
    #pragma unroll
    for (int e = 0; e < NEXP; ++e) { p[e] = __expf(l[e] - mx); s += p[e]; }
    float inv = 1.0f / s;
    #pragma unroll
    for (int e = 0; e < NEXP; ++e) p[e] *= inv;
    int i0 = 0;
    #pragma unroll
    for (int e = 1; e < NEXP; ++e) if (p[e] > p[i0]) i0 = e;
    int i1 = (i0 == 0) ? 1 : 0;
    #pragma unroll
    for (int e = 0; e < NEXP; ++e) if (e != i0 && p[e] > p[i1]) i1 = e;
    float r0 = p[i0], r1 = p[i1];
    float rs = 1.0f / (r0 + r1);
    r0 *= rs; r1 *= rs;
    #pragma unroll
    for (int e = 0; e < NEXP; ++e)
        we[n * NEXP + e] = (e == i0) ? r0 : ((e == i1) ? r1 : 0.0f);
    #pragma unroll
    for (int e = 0; e < NEXP; ++e) atomicAdd(&psum[e], p[e]);
    atomicAdd(&cnt[i0], 1.0f);
    atomicAdd(&cnt[i1], 1.0f);
}

// ============================================================================
// LDS-staged double-buffered bf16 GEMM (block 128x128, wave tile 32x64).
// A: [M,K] bf16 row-major, B: [Nn,K] bf16 row-major, C = A . B^T.
// 256 threads = 8 waves (4 M x 2 N). K-step 32. Async global->LDS pipeline.
// MODE 0: C = bf16(relu(acc)^2) -> Cbf   (GEMM1)
// MODE 1: Cf += we[m] * acc               (GEMM2, f32 accumulate into output)
// ============================================================================
template<int MODE>
__global__ __launch_bounds__(256) void moe_gemm_kernel(
    const unsigned short* __restrict__ A, const unsigned short* __restrict__ B,
    unsigned short* __restrict__ Cbf, float* __restrict__ Cf,
    const float* __restrict__ weE, int Nn, int K) {

    __shared__ __attribute__((aligned(16))) unsigned short shA[2][128 * PADE];
    __shared__ __attribute__((aligned(16))) unsigned short shB[2][128 * PADE];

    const int t    = threadIdx.x;
    const int lane = t & 31;
    const int w    = t >> 5;
    const int tmw  = (w >> 1) * 32;   // wave M offset in block tile
    const int tnw  = (w & 1) * 64;    // wave N offset in block tile
    const int tmB  = blockIdx.y * 128;
    const int tnB  = blockIdx.x * 128;
    const unsigned short* Ab = A + (size_t)tmB * K;
    const unsigned short* Bb = B + (size_t)tnB * K;
    const unsigned ldsA0 = (unsigned)(size_t)&shA[0][0];
    const unsigned ldsA1 = (unsigned)(size_t)&shA[1][0];
    const unsigned ldsB0 = (unsigned)(size_t)&shB[0][0];
    const unsigned ldsB1 = (unsigned)(size_t)&shB[1][0];
    const int crow = t >> 2, cq = t & 3;  // copy mapping: rows crow, crow+64; 16B chunk cq

    v8f acc[2][4] = {};
    const int nk = K / 32;

    // prologue: stage 0 into buffer 0
    async_copy16(ldsA0 + (unsigned)(crow * ROWB + cq * 16),
                 Ab + (size_t)crow * K + cq * 8);
    async_copy16(ldsA0 + (unsigned)((crow + 64) * ROWB + cq * 16),
                 Ab + (size_t)(crow + 64) * K + cq * 8);
    async_copy16(ldsB0 + (unsigned)(crow * ROWB + cq * 16),
                 Bb + (size_t)crow * K + cq * 8);
    async_copy16(ldsB0 + (unsigned)((crow + 64) * ROWB + cq * 16),
                 Bb + (size_t)(crow + 64) * K + cq * 8);

    const int m16 = lane & 15, hi = lane >> 4;

    for (int i = 0; i < nk; ++i) {
        const int cur = i & 1;
        __syncthreads();  // everyone finished reading buf[cur^1] from iter i-1
        if (i + 1 < nk) { // stage i+1 into the other buffer
            const int kb = (i + 1) * 32;
            const unsigned la = (cur == 0) ? ldsA1 : ldsA0;
            const unsigned lb = (cur == 0) ? ldsB1 : ldsB0;
            async_copy16(la + (unsigned)(crow * ROWB + cq * 16),
                         Ab + (size_t)crow * K + kb + cq * 8);
            async_copy16(la + (unsigned)((crow + 64) * ROWB + cq * 16),
                         Ab + (size_t)(crow + 64) * K + kb + cq * 8);
            async_copy16(lb + (unsigned)(crow * ROWB + cq * 16),
                         Bb + (size_t)crow * K + kb + cq * 8);
            async_copy16(lb + (unsigned)((crow + 64) * ROWB + cq * 16),
                         Bb + (size_t)(crow + 64) * K + kb + cq * 8);
            wait_async_le4();  // in-order completion: stage i's 4 copies have landed
        } else {
            wait_async_0();
        }
        __syncthreads();  // all waves' stage-i copies visible in LDS

        const unsigned short* sA = &shA[cur][0];
        const unsigned short* sB = &shB[cur][0];
        v16bf a0 = frag_from_lds(sA + (tmw + m16) * PADE + hi * 8);
        v16bf a1 = frag_from_lds(sA + (tmw + 16 + m16) * PADE + hi * 8);
        v16bf b0 = frag_from_lds(sB + (tnw + m16) * PADE + hi * 8);
        v16bf b1 = frag_from_lds(sB + (tnw + 16 + m16) * PADE + hi * 8);
        v16bf b2 = frag_from_lds(sB + (tnw + 32 + m16) * PADE + hi * 8);
        v16bf b3 = frag_from_lds(sB + (tnw + 48 + m16) * PADE + hi * 8);
        acc[0][0] = __builtin_amdgcn_wmma_f32_16x16x32_bf16(false, a0, false, b0, (short)0, acc[0][0], false, false);
        acc[0][1] = __builtin_amdgcn_wmma_f32_16x16x32_bf16(false, a0, false, b1, (short)0, acc[0][1], false, false);
        acc[0][2] = __builtin_amdgcn_wmma_f32_16x16x32_bf16(false, a0, false, b2, (short)0, acc[0][2], false, false);
        acc[0][3] = __builtin_amdgcn_wmma_f32_16x16x32_bf16(false, a0, false, b3, (short)0, acc[0][3], false, false);
        acc[1][0] = __builtin_amdgcn_wmma_f32_16x16x32_bf16(false, a1, false, b0, (short)0, acc[1][0], false, false);
        acc[1][1] = __builtin_amdgcn_wmma_f32_16x16x32_bf16(false, a1, false, b1, (short)0, acc[1][1], false, false);
        acc[1][2] = __builtin_amdgcn_wmma_f32_16x16x32_bf16(false, a1, false, b2, (short)0, acc[1][2], false, false);
        acc[1][3] = __builtin_amdgcn_wmma_f32_16x16x32_bf16(false, a1, false, b3, (short)0, acc[1][3], false, false);
    }

    // epilogue (C layout: row m = base + 8*hi + r, col n = base + m16)
    if constexpr (MODE == 0) {
        #pragma unroll
        for (int ii = 0; ii < 2; ++ii) {
            #pragma unroll
            for (int jj = 0; jj < 4; ++jj) {
                #pragma unroll
                for (int r = 0; r < 8; ++r) {
                    int m = tmB + tmw + ii * 16 + hi * 8 + r;
                    int n = tnB + tnw + jj * 16 + m16;
                    float v = acc[ii][jj][r];
                    v = v > 0.0f ? v * v : 0.0f;          // relu(z)^2
                    Cbf[(size_t)m * Nn + n] = f32_to_bf16_rne(v);
                }
            }
        }
    } else {
        #pragma unroll
        for (int ii = 0; ii < 2; ++ii) {
            #pragma unroll
            for (int r = 0; r < 8; ++r) {
                int m = tmB + tmw + ii * 16 + hi * 8 + r;
                float wv = weE[(size_t)m * NEXP];   // weE pre-offset by expert index
                #pragma unroll
                for (int jj = 0; jj < 4; ++jj) {
                    int n = tnB + tnw + jj * 16 + m16;
                    size_t idx = (size_t)m * Nn + n;
                    Cf[idx] += wv * acc[ii][jj][r];  // serialized expert launches: race-free
                }
            }
        }
    }
}

// ---------- balance loss ----------
__global__ void moe_loss_kernel(const float* __restrict__ psum,
                                const float* __restrict__ cnt,
                                float* __restrict__ lossOut) {
    if (threadIdx.x == 0 && blockIdx.x == 0) {
        const float invN = 1.0f / (float)N_TOK;
        float l = 0.0f;
        #pragma unroll
        for (int e = 0; e < NEXP; ++e) l += (psum[e] * invN) * (cnt[e] * invN);
        *lossOut = l * (float)NEXP;
    }
}

extern "C" void kernel_launch(void* const* d_in, const int* in_sizes, int n_in,
                              void* d_out, int out_size, void* d_ws, size_t ws_size,
                              hipStream_t stream) {
    const float* x      = (const float*)d_in[0];   // [N_TOK, DIM]
    const float* Wgate  = (const float*)d_in[1];   // [NEXP, DIM]
    const float* Wfc    = (const float*)d_in[2];   // [NEXP, HID, DIM]
    const float* Wproj  = (const float*)d_in[3];   // [NEXP, DIM, HID]
    float* out = (float*)d_out;                    // [N_TOK*DIM] ++ [1] loss

    // workspace layout (all regions well aligned by construction)
    char* ws = (char*)d_ws;
    size_t off = 0;
    unsigned short* xb    = (unsigned short*)(ws + off); off += (size_t)N_TOK * DIM * 2;        // 8 MB
    unsigned short* wfcb  = (unsigned short*)(ws + off); off += (size_t)NEXP * HID * DIM * 2;   // 67 MB
    unsigned short* wprjb = (unsigned short*)(ws + off); off += (size_t)NEXP * DIM * HID * 2;   // 67 MB
    unsigned short* hb    = (unsigned short*)(ws + off); off += (size_t)N_TOK * HID * 2;        // 33.5 MB
    float* logits = (float*)(ws + off); off += (size_t)N_TOK * NEXP * 4;
    float* we     = (float*)(ws + off); off += (size_t)N_TOK * NEXP * 4;
    float* psum   = (float*)(ws + off); off += 256;
    float* cnt    = (float*)(ws + off); off += 256;

    const int nOut = N_TOK * DIM;

    // 1) zero accumulators / output
    moe_zero_kernel<<<(nOut + 255) / 256, 256, 0, stream>>>(out, psum, cnt, nOut);

    // 2) fp32 -> bf16 conversions
    {
        int n4 = (N_TOK * DIM) / 4;
        moe_cvt_bf16_kernel<<<(n4 + 255) / 256, 256, 0, stream>>>(x, xb, n4);
        n4 = (NEXP * HID * DIM) / 4;
        moe_cvt_bf16_kernel<<<(n4 + 255) / 256, 256, 0, stream>>>(Wfc, wfcb, n4);
        n4 = (NEXP * DIM * HID) / 4;
        moe_cvt_bf16_kernel<<<(n4 + 255) / 256, 256, 0, stream>>>(Wproj, wprjb, n4);
    }

    // 3) gating
    moe_gate_logits_kernel<<<(N_TOK * NEXP + 255) / 256, 256, 0, stream>>>(x, Wgate, logits);
    moe_gate_finalize_kernel<<<(N_TOK + 255) / 256, 256, 0, stream>>>(logits, we, psum, cnt);

    // 4) per-expert dense MLP, masked-combined by routing weight (same math as reference)
    dim3 g1(HID / 128, N_TOK / 128);   // (32, 32)
    dim3 g2(DIM / 128, N_TOK / 128);   // (8, 32)
    for (int e = 0; e < NEXP; ++e) {
        const unsigned short* wfe = wfcb  + (size_t)e * HID * DIM;
        const unsigned short* wpe = wprjb + (size_t)e * DIM * HID;
        moe_gemm_kernel<0><<<g1, 256, 0, stream>>>(xb, wfe, hb, nullptr, nullptr, HID, DIM);
        moe_gemm_kernel<1><<<g2, 256, 0, stream>>>(hb, wpe, nullptr, out, we + e, DIM, HID);
    }

    // 5) balance loss scalar
    moe_loss_kernel<<<1, 1, 0, stream>>>(psum, cnt, out + (size_t)N_TOK * DIM);
}
